// GraphAttentionEncoder_68264210202884
// MI455X (gfx1250) — matrix operand order repaired and checked by reading, not compiled, not run on hardware
//
#include <hip/hip_runtime.h>
#include <hip/hip_bf16.h>

#define N_NODES  50000
#define N_EDGES  1200000
#define D_MODEL  64
#define N_HEADS  4
#define D_HEAD   16
#define NUM_BASIS 41
#define WT_LD    72   // padded LDS stride (halves) to spread banks

typedef __attribute__((ext_vector_type(16))) _Float16 v16h;
typedef __attribute__((ext_vector_type(8)))  _Float16 v8h;
typedef __attribute__((ext_vector_type(8)))  float    v8f;

// ---------------- helpers ----------------

__device__ __forceinline__ unsigned flip_f32(float f) {
  unsigned u = __float_as_uint(f);
  return (u & 0x80000000u) ? ~u : (u | 0x80000000u);
}
__device__ __forceinline__ float unflip_f32(unsigned u) {
  return (u & 0x80000000u) ? __uint_as_float(u & 0x7FFFFFFFu) : __uint_as_float(~u);
}

// Stage a row-major f32 weight W[rows x cols] into LDS transposed+padded:
// wt[c*WT_LD + k] = (f16)W[k*cols + c], rows..63 zero-filled. Call with all threads, then sync.
__device__ __forceinline__ void stage_wt(const float* __restrict__ W, int rows, int cols,
                                         _Float16* wt, int t, int nthreads) {
  for (int idx = t; idx < rows * cols; idx += nthreads) {
    int c = idx % cols, k = idx / cols;
    wt[c * WT_LD + k] = (_Float16)W[idx];
  }
  for (int idx = t; idx < (64 - rows) * cols; idx += nthreads) {
    int c = idx % cols, k = rows + idx / cols;
    wt[c * WT_LD + k] = (_Float16)0.f;
  }
}

// A fragment (16x32 f16) from an LDS tile hn[16][64]; two aligned 16B reads.
// ISA layout: lanes 0-15: e0..7->K kb+0..7, e8..15->K kb+16..23 ; lanes 16-31: +8/+24.
__device__ __forceinline__ v16h load_a_frag_lds(const _Float16* hn, int lane, int kb) {
  int r = lane & 15, hf = lane >> 4;
  v8h lo = *(const v8h*)(hn + r * 64 + kb + hf * 8);
  v8h hi = *(const v8h*)(hn + r * 64 + kb + 16 + hf * 8);
  v16h a;
#pragma unroll
  for (int e = 0; e < 8; ++e) { a[e] = lo[e]; a[8 + e] = hi[e]; }
  return a;
}

// B fragment (32x16 f16) from transposed LDS weights wt[col][k]; two aligned 16B reads.
// ISA layout: lanes 0-15: e0..7->K kb+0..7, e8..15->K kb+16..23 ; lanes 16-31: +8/+24; col = lane&15.
__device__ __forceinline__ v16h load_b_frag_lds(const _Float16* wt, int ncol, int kb, int lane) {
  int n = lane & 15, hf = lane >> 4;
  const _Float16* p = wt + (ncol + n) * WT_LD + kb + hf * 8;
  v8h lo = *(const v8h*)(p);
  v8h hi = *(const v8h*)(p + 16);
  v16h b;
#pragma unroll
  for (int e = 0; e < 8; ++e) { b[e] = lo[e]; b[8 + e] = hi[e]; }
  return b;
}

// ---------------- kernels ----------------

__global__ void zero_kernel(unsigned* __restrict__ p, int n) {
  int i = blockIdx.x * blockDim.x + threadIdx.x;
  if (i < n) p[i] = 0u;
}

// h = embed_table[x] * sqrt(64)
__global__ void embed_kernel(const int* __restrict__ x, const float* __restrict__ tab,
                             float* __restrict__ h) {
  int i = blockIdx.x * blockDim.x + threadIdx.x;          // over N_NODES*64
  int node = i >> 6, c = i & 63;
  h[i] = tab[x[node] * 64 + c] * 8.0f;
}

// e_attr[e, :] = (rbf_basis(dist_e) @ W_edge + b_edge)  -> f16, via WMMA.
// basis[e,k] = exp(-(dist*5 - k)^2)/1.12; k>=41 handled by zero rows of staged W.
__global__ __launch_bounds__(128) void edge_attr_kernel(const float* __restrict__ dist,
                                                        const float* __restrict__ We,
                                                        const float* __restrict__ be,
                                                        _Float16* __restrict__ eattr) {
  __shared__ __align__(16) _Float16 wt[64 * WT_LD];
  __shared__ float bias[64];
  int t = threadIdx.x, wave = t >> 5, lane = t & 31;
  stage_wt(We, NUM_BASIS, 64, wt, t, 128);
  if (t < 64) bias[t] = be[t];
  __syncthreads();

  int ebase = blockIdx.x * 64 + wave * 16;
  float d5 = dist[ebase + (lane & 15)] * 5.0f;
  int hf = lane >> 4, n = lane & 15;

  v16h a[2];
#pragma unroll
  for (int ks = 0; ks < 2; ++ks) {
#pragma unroll
    for (int e = 0; e < 16; ++e) {
      int k = ks * 32 + ((e < 8) ? (hf * 8 + e) : (16 + hf * 8 + (e - 8)));
      float df = d5 - (float)k;
      // fast exp: arg <= 0, underflow->0 is correct; k>=41 killed by zero B rows
      a[ks][e] = (_Float16)(__expf(-df * df) * (1.0f / 1.12f));
    }
  }
#pragma unroll
  for (int nt = 0; nt < 4; ++nt) {
    v8f acc = {};
#pragma unroll
    for (int ks = 0; ks < 2; ++ks) {
      v16h b = load_b_frag_lds(wt, nt * 16, ks * 32, lane);
      acc = __builtin_amdgcn_wmma_f32_16x16x32_f16(false, a[ks], false, b,
                                                   (short)0, acc, false, false);
    }
#pragma unroll
    for (int g = 0; g < 8; ++g) {
      int M = g + hf * 8, col = nt * 16 + n;
      eattr[(size_t)(ebase + M) * 64 + col] = (_Float16)(acc[g] + bias[col]);
    }
  }
}

// Fused LayerNorm + QKV projection (50000x64 @ 64x192), WMMA, f16 output.
__global__ __launch_bounds__(128) void ln_qkv_kernel(const float* __restrict__ h,
                                                     const float* __restrict__ Wq,
                                                     const float* __restrict__ bq,
                                                     const float* __restrict__ lns,
                                                     const float* __restrict__ lnb,
                                                     _Float16* __restrict__ qkv) {
  __shared__ __align__(16) _Float16 wt[192 * WT_LD];
  __shared__ __align__(16) _Float16 hn[16 * 64];
  int t = threadIdx.x;
  stage_wt(Wq, 64, 192, wt, t, 128);

  int base = blockIdx.x * 16;
  int r = t >> 3, c8 = (t & 7) * 8;
  const float* hrow = h + (size_t)(base + r) * 64;
  float v[8], s0 = 0.f, s1 = 0.f;
#pragma unroll
  for (int i = 0; i < 8; ++i) { v[i] = hrow[c8 + i]; s0 += v[i]; s1 += v[i] * v[i]; }
#pragma unroll
  for (int d = 4; d > 0; d >>= 1) { s0 += __shfl_down(s0, d, 8); s1 += __shfl_down(s1, d, 8); }
  float mu = __shfl(s0, 0, 8) * (1.0f / 64.0f);
  float ms = __shfl(s1, 0, 8) * (1.0f / 64.0f);
  float rstd = rsqrtf(ms - mu * mu + 1e-5f);
#pragma unroll
  for (int i = 0; i < 8; ++i) {
    int c = c8 + i;
    hn[r * 64 + c] = (_Float16)(((v[i] - mu) * rstd) * lns[c] + lnb[c]);
  }
  __syncthreads();

  int wave = t >> 5, lane = t & 31;
  int n = lane & 15, hf = lane >> 4;
#pragma unroll
  for (int i = 0; i < 3; ++i) {
    int nc = (wave * 3 + i) * 16;
    v8f acc = {};
#pragma unroll
    for (int ks = 0; ks < 2; ++ks) {
      v16h a = load_a_frag_lds(hn, lane, ks * 32);
      v16h b = load_b_frag_lds(wt, nc, ks * 32, lane);
      acc = __builtin_amdgcn_wmma_f32_16x16x32_f16(false, a, false, b,
                                                   (short)0, acc, false, false);
    }
#pragma unroll
    for (int g = 0; g < 8; ++g) {
      int M = g + hf * 8, col = nc + n;
      qkv[(size_t)(base + M) * 192 + col] = (_Float16)(acc[g] + bq[col]);
    }
  }
}

// Per (edge, head): logit = 0.25 * dot(q[dst], k[src] + e_attr); atomicMax of flipped bits.
__global__ __launch_bounds__(256) void logit_kernel(const int* __restrict__ ei,
                                                    const _Float16* __restrict__ qkv,
                                                    const _Float16* __restrict__ eattr,
                                                    float* __restrict__ logits,
                                                    unsigned* __restrict__ mflip) {
  int tid = blockIdx.x * blockDim.x + threadIdx.x;       // over N_EDGES*4
  int e = tid >> 2, hh = tid & 3;
  int src = ei[e], dst = ei[N_EDGES + e];
  const v8h* qp = (const v8h*)(qkv + (size_t)dst * 192 + hh * 16);
  const v8h* kp = (const v8h*)(qkv + (size_t)src * 192 + 64 + hh * 16);
  const v8h* ep = (const v8h*)(eattr + (size_t)e * 64 + hh * 16);
  float acc = 0.f;
#pragma unroll
  for (int j = 0; j < 2; ++j) {
    v8h q = qp[j], k = kp[j], ea = ep[j];
#pragma unroll
    for (int d = 0; d < 8; ++d) acc += (float)q[d] * ((float)k[d] + (float)ea[d]);
  }
  acc *= 0.25f;
  logits[tid] = acc;
  atomicMax(&mflip[dst * 4 + hh], flip_f32(acc));
}

// Per (edge, head): w = exp(logit - max); s[dst] += w; agg[dst] += w * v[src].
__global__ __launch_bounds__(256) void agg_kernel(const int* __restrict__ ei,
                                                  const _Float16* __restrict__ qkv,
                                                  const float* __restrict__ logits,
                                                  const unsigned* __restrict__ mflip,
                                                  float* __restrict__ ssum,
                                                  float* __restrict__ agg) {
  int tid = blockIdx.x * blockDim.x + threadIdx.x;       // over N_EDGES*4
  int e = tid >> 2, hh = tid & 3;
  int src = ei[e], dst = ei[N_EDGES + e];
  float m = unflip_f32(mflip[dst * 4 + hh]);
  float w = __expf(logits[tid] - m);   // arg <= 0 by construction
  atomicAdd(&ssum[dst * 4 + hh], w);
  const v8h* vp = (const v8h*)(qkv + (size_t)src * 192 + 128 + hh * 16);
#pragma unroll
  for (int j = 0; j < 2; ++j) {
    v8h vv = vp[j];
#pragma unroll
    for (int d = 0; d < 8; ++d)
      atomicAdd(&agg[(size_t)dst * 64 + hh * 16 + j * 8 + d], w * (float)vv[d]);
  }
}

// out = h + (agg / (s + 1e-16)) @ W_out + b_out, WMMA fused with normalization.
__global__ __launch_bounds__(128) void out_kernel(const float* __restrict__ h_in,
                                                  const float* __restrict__ agg,
                                                  const float* __restrict__ ssum,
                                                  const float* __restrict__ Wo,
                                                  const float* __restrict__ bo,
                                                  float* __restrict__ out) {
  __shared__ __align__(16) _Float16 wt[64 * WT_LD];
  __shared__ __align__(16) _Float16 an[16 * 64];
  int t = threadIdx.x;
  stage_wt(Wo, 64, 64, wt, t, 128);

  int base = blockIdx.x * 16;
  int r = t >> 3, c8 = (t & 7) * 8;
  const float* arow = agg + (size_t)(base + r) * 64;
  const float* srow = ssum + (size_t)(base + r) * 4;
#pragma unroll
  for (int i = 0; i < 8; ++i) {
    int c = c8 + i;
    an[r * 64 + c] = (_Float16)(arow[c] / (srow[c >> 4] + 1e-16f));
  }
  __syncthreads();

  int wave = t >> 5, lane = t & 31;
  int n = lane & 15, hf = lane >> 4;
  int nc = wave * 16;
  v8f acc = {};
#pragma unroll
  for (int ks = 0; ks < 2; ++ks) {
    v16h a = load_a_frag_lds(an, lane, ks * 32);
    v16h b = load_b_frag_lds(wt, nc, ks * 32, lane);
    acc = __builtin_amdgcn_wmma_f32_16x16x32_f16(false, a, false, b,
                                                 (short)0, acc, false, false);
  }
#pragma unroll
  for (int g = 0; g < 8; ++g) {
    int M = g + hf * 8, col = nc + n;
    size_t idx = (size_t)(base + M) * 64 + col;
    out[idx] = h_in[idx] + acc[g] + bo[col];
  }
}

// ---------------- launcher ----------------

extern "C" void kernel_launch(void* const* d_in, const int* in_sizes, int n_in,
                              void* d_out, int out_size, void* d_ws, size_t ws_size,
                              hipStream_t stream) {
  (void)in_sizes; (void)n_in; (void)out_size; (void)ws_size;
  const int*   x      = (const int*)d_in[0];
  const int*   ei     = (const int*)d_in[1];
  const float* dist   = (const float*)d_in[2];
  const float* embed  = (const float*)d_in[3];
  const float* W_edge = (const float*)d_in[4];
  const float* b_edge = (const float*)d_in[5];
  const float* ln_s   = (const float*)d_in[6];
  const float* ln_b   = (const float*)d_in[7];
  const float* W_qkv  = (const float*)d_in[8];
  const float* b_qkv  = (const float*)d_in[9];
  const float* W_out  = (const float*)d_in[10];
  const float* b_out  = (const float*)d_in[11];
  float* out = (float*)d_out;

  char* ws = (char*)d_ws;
  size_t off = 0;
  auto take = [&](size_t bytes) -> char* {
    char* p = ws + off;
    off = (off + bytes + 255) & ~(size_t)255;
    return p;
  };
  float*    h      = (float*)   take((size_t)N_NODES * 64 * 4);
  _Float16* qkv    = (_Float16*)take((size_t)N_NODES * 192 * 2);
  float*    logits = (float*)   take((size_t)N_EDGES * 4 * 4);
  unsigned* mflip  = (unsigned*)take((size_t)N_NODES * 4 * 4);
  float*    ssum   = (float*)   take((size_t)N_NODES * 4 * 4);
  float*    agg    = (float*)   take((size_t)N_NODES * 64 * 4);
  _Float16* eattr  = (_Float16*)take((size_t)N_EDGES * 64 * 2);

  embed_kernel<<<(N_NODES * 64) / 256, 256, 0, stream>>>(x, embed, h);
  edge_attr_kernel<<<N_EDGES / 64, 128, 0, stream>>>(dist, W_edge, b_edge, eattr);

  for (int l = 0; l < 2; ++l) {
    zero_kernel<<<(N_NODES * 4 + 255) / 256, 256, 0, stream>>>(mflip, N_NODES * 4);
    zero_kernel<<<(N_NODES * 4 + 255) / 256, 256, 0, stream>>>((unsigned*)ssum, N_NODES * 4);
    zero_kernel<<<(N_NODES * 64 + 255) / 256, 256, 0, stream>>>((unsigned*)agg, N_NODES * 64);

    ln_qkv_kernel<<<N_NODES / 16, 128, 0, stream>>>(h, W_qkv + (size_t)l * 64 * 192,
                                                    b_qkv + l * 192, ln_s + l * 64,
                                                    ln_b + l * 64, qkv);
    logit_kernel<<<(N_EDGES * 4) / 256, 256, 0, stream>>>(ei, qkv, eattr, logits, mflip);
    agg_kernel<<<(N_EDGES * 4) / 256, 256, 0, stream>>>(ei, qkv, logits, mflip, ssum, agg);
    out_kernel<<<N_NODES / 16, 128, 0, stream>>>(h, agg, ssum, W_out + (size_t)l * 4096,
                                                 b_out + l * 64, (l == 0) ? h : out);
  }
}
